// GIN_DGL_84851373900195
// MI455X (gfx1250) — matrix operand order repaired and checked by reading, not compiled, not run on hardware
//
#include <hip/hip_runtime.h>
#include <hip/hip_bf16.h>

#define D 256
#define NNODES 50000
#define NEDGES 800000
#define NGRAPH 512
#define NCLASS 10
#define BN_EPS 1e-5f
#define ND (NNODES * D)          // 12,800,000
#define MATSZ (D * D)            // 65,536 elements per weight matrix

typedef __attribute__((ext_vector_type(16))) __bf16 v16bf;
typedef __attribute__((ext_vector_type(8)))  float  v8f;

union BFrag { v16bf v; uint4 q[2]; };
union F8    { v8f  v; float f[8]; };

__device__ __forceinline__ unsigned short f2bf(float f) {
  unsigned int u = __float_as_uint(f);
  unsigned int r = u + 0x7FFFu + ((u >> 16) & 1u);   // round-to-nearest-even
  return (unsigned short)(r >> 16);
}

// ---------------------------------------------------------------- utilities
__global__ void fill_kernel(float* p, long long n) {
  long long i = (long long)blockIdx.x * blockDim.x + threadIdx.x;
  if (i < n) p[i] = 0.0f;
}

// Pack fp32 weights (4x GIN W, then fc1_w) into the per-lane bf16 B-fragment
// layout: [mat][ntile(16)][kt(8)][lane(32)][j(16)],  K = kt*32+(lane>>4)*16+j,
// N = ntile*16 + (lane&15).  Each GEMM lane then loads 32 contiguous bytes.
__global__ void pack_weights_kernel(const float* Ws, const float* fc1_w,
                                    unsigned short* Wp) {
  int gid = blockIdx.x * blockDim.x + threadIdx.x;
  if (gid >= 5 * MATSZ) return;
  int m = gid >> 16;
  int r = gid & 65535;
  int ntile = r >> 12;
  int kt    = (r >> 9) & 7;
  int lane  = (r >> 4) & 31;
  int j     = r & 15;
  int k = kt * 32 + (lane >> 4) * 16 + j;
  int n = (ntile << 4) + (lane & 15);
  float val = (m < 4) ? Ws[m * MATSZ + k * D + n] : fc1_w[k * D + n];
  Wp[gid] = f2bf(val);
}

// ---------------------------------------------------------------- GNN steps
// agg[dst[e]] += h[src[e]]   (64 threads per edge, float4 gather, 4 atomics)
__global__ void scatter_kernel(const float* __restrict__ h,
                               const int* __restrict__ src,
                               const int* __restrict__ dst,
                               float* __restrict__ agg) {
  long long gid = (long long)blockIdx.x * blockDim.x + threadIdx.x;
  int e    = (int)(gid >> 6);
  int part = (int)(gid & 63);
  int s = src[e], d = dst[e];
  float4 v = *(const float4*)(h + (size_t)s * D + part * 4);
  float* ap = agg + (size_t)d * D + part * 4;
  atomicAdd(ap + 0, v.x); atomicAdd(ap + 1, v.y);
  atomicAdd(ap + 2, v.z); atomicAdd(ap + 3, v.w);
}

// xb = bf16(h + agg)
__global__ void addcvt_kernel(const float* __restrict__ h,
                              const float* __restrict__ agg,
                              unsigned short* __restrict__ xb, int n) {
  int idx = blockIdx.x * blockDim.x + threadIdx.x;
  if (idx >= n) return;
  xb[idx] = f2bf(h[idx] + agg[idx]);
}

// out[M x 256] = xb[M x 256] @ Wpacked + bias.
// One block = 64 rows x 256 cols.  4 waves; each wave owns 4 N-tiles and all
// 4 M-tiles (4x4 grid of 16x16 accumulators), so every B fragment load feeds
// 4 WMMAs (4x less L2 weight traffic).  8 K-steps of v_wmma_f32_16x16x32_bf16.
// OOB rows are zero-padded in LDS so EXEC stays all-ones around the WMMAs.
__global__ __launch_bounds__(128)
void wmma_gemm_kernel(const unsigned short* __restrict__ xb,
                      const unsigned short* __restrict__ wp,
                      const float* __restrict__ bias,
                      float* __restrict__ out, int M) {
  __shared__ uint4 Aq[2048];                     // 64 x 256 bf16 = 32 KB
  const int m0 = blockIdx.x * 64;
  const uint4 zero4 = make_uint4(0u, 0u, 0u, 0u);
  // Cooperative tile load: 2048 uint4 chunks, 32 chunks per row.
  for (int c = threadIdx.x; c < 2048; c += 128) {
    int row = c >> 5;
    Aq[c] = (m0 + row < M)
              ? *(const uint4*)(xb + (size_t)(m0 + row) * D + ((c & 31) << 3))
              : zero4;
  }
  __syncthreads();

  const int wave = threadIdx.x >> 5;
  const int lane = threadIdx.x & 31;
  const int half = lane >> 4;
  const int lcol = lane & 15;
  const unsigned short* As = (const unsigned short*)Aq;

  F8 acc[4][4];                                  // [mtile][ntile]
#pragma unroll
  for (int mt = 0; mt < 4; ++mt)
#pragma unroll
    for (int nt = 0; nt < 4; ++nt) acc[mt][nt].v = (v8f){0, 0, 0, 0, 0, 0, 0, 0};

#pragma unroll
  for (int kt = 0; kt < 8; ++kt) {
    const int kb = kt * 32 + half * 16;          // A: K = kb..kb+15
    BFrag a[4];
#pragma unroll
    for (int mt = 0; mt < 4; ++mt) {
      const unsigned short* ap = As + (size_t)(mt * 16 + lcol) * D + kb;
      a[mt].q[0] = *(const uint4*)(ap);
      a[mt].q[1] = *(const uint4*)(ap + 8);
    }
#pragma unroll
    for (int nt = 0; nt < 4; ++nt) {
      const int ntile = wave * 4 + nt;
      BFrag b;
      const unsigned short* bp = wp + (size_t)(ntile * 8 + kt) * 512 + lane * 16;
      b.q[0] = *(const uint4*)(bp);
      b.q[1] = *(const uint4*)(bp + 8);
#pragma unroll
      for (int mt = 0; mt < 4; ++mt) {
        acc[mt][nt].v = __builtin_amdgcn_wmma_f32_16x16x32_bf16(
            false, a[mt].v, false, b.v, (short)0, acc[mt][nt].v, false, false);
      }
    }
  }

#pragma unroll
  for (int nt = 0; nt < 4; ++nt) {
    const int col = (wave * 4 + nt) * 16 + lcol;
    const float bsv = bias[col];
#pragma unroll
    for (int mt = 0; mt < 4; ++mt) {
#pragma unroll
      for (int r = 0; r < 8; ++r) {
        const int rowg = m0 + mt * 16 + r + half * 8;
        if (rowg < M) out[(size_t)rowg * D + col] = acc[mt][nt].f[r] + bsv;
      }
    }
  }
}

// per-column sum / sum-of-squares partials (coalesced rows, atomic merge)
__global__ void colstats_kernel(const float* __restrict__ z,
                                float* __restrict__ stats, int nrows) {
  int col = threadIdx.x;
  int r0 = blockIdx.x * 256;
  int r1 = r0 + 256; if (r1 > nrows) r1 = nrows;
  float s = 0.f, s2 = 0.f;
  for (int r = r0; r < r1; ++r) {
    float v = z[(size_t)r * D + col];
    s += v; s2 += v * v;
  }
  atomicAdd(&stats[col], s);
  atomicAdd(&stats[D + col], s2);
}

__global__ void bnrelu_kernel(const float* __restrict__ z,
                              const float* __restrict__ stats,
                              const float* __restrict__ gamma,
                              const float* __restrict__ beta,
                              float* __restrict__ out, int n) {
  int idx = blockIdx.x * blockDim.x + threadIdx.x;
  if (idx >= n) return;
  int col = idx & (D - 1);
  float mean = stats[col] * (1.0f / NNODES);
  float var  = stats[D + col] * (1.0f / NNODES) - mean * mean;
  float inv  = rsqrtf(var + BN_EPS);
  float v = gamma[col] * (z[idx] - mean) * inv + beta[col];
  out[idx] = v > 0.f ? v : 0.f;
}

// ---------------------------------------------------------------- pooling
__global__ void count_kernel(const int* __restrict__ gids,
                             float* __restrict__ counts, int n) {
  int i = blockIdx.x * blockDim.x + threadIdx.x;
  if (i < n) atomicAdd(&counts[gids[i]], 1.0f);
}

__global__ void poolsum_kernel(const float* __restrict__ h,
                               const int* __restrict__ gids,
                               float* __restrict__ pool, int n) {
  int idx = blockIdx.x * blockDim.x + threadIdx.x;
  if (idx >= n) return;
  int g = gids[idx >> 8];
  atomicAdd(&pool[(size_t)g * D + (idx & (D - 1))], h[idx]);
}

__global__ void poolfin_kernel(const float* __restrict__ pool,
                               const float* __restrict__ counts,
                               unsigned short* __restrict__ hgb) {
  int idx = blockIdx.x * blockDim.x + threadIdx.x;   // G*D threads
  float c = counts[idx >> 8];
  float v = pool[idx] / fmaxf(c, 1.0f);
  hgb[idx] = f2bf(v);
}

// ---------------------------------------------------------------- head
__global__ void elu_kernel(const float* __restrict__ z,
                           float* __restrict__ out, int n) {
  int idx = blockIdx.x * blockDim.x + threadIdx.x;
  if (idx >= n) return;
  float v = z[idx];
  out[idx] = v > 0.f ? v : (expf(v) - 1.0f);
}

__global__ void fc2_kernel(const float* __restrict__ x,
                           const float* __restrict__ w,
                           const float* __restrict__ b,
                           float* __restrict__ logits) {
  int idx = blockIdx.x * blockDim.x + threadIdx.x;
  if (idx >= NGRAPH * NCLASS) return;
  int g = idx / NCLASS, c = idx % NCLASS;
  float s = b[c];
  const float* xr = x + (size_t)g * D;
  for (int k = 0; k < D; ++k) s += xr[k] * w[k * NCLASS + c];
  logits[idx] = s;
}

// log_softmax over dim 0 (graphs). One block per class, 256 threads x 2 rows.
__global__ void logsoftmax_kernel(const float* __restrict__ logits,
                                  float* __restrict__ out) {
  __shared__ float red[256];
  int c = blockIdx.x;
  int t = threadIdx.x;
  float v0 = logits[t * NCLASS + c];
  float v1 = logits[(t + 256) * NCLASS + c];
  red[t] = fmaxf(v0, v1);
  __syncthreads();
  for (int s = 128; s > 0; s >>= 1) {
    if (t < s) red[t] = fmaxf(red[t], red[t + s]);
    __syncthreads();
  }
  float mx = red[0];
  __syncthreads();
  red[t] = expf(v0 - mx) + expf(v1 - mx);
  __syncthreads();
  for (int s = 128; s > 0; s >>= 1) {
    if (t < s) red[t] += red[t + s];
    __syncthreads();
  }
  float lse = logf(red[0]) + mx;
  out[t * NCLASS + c] = v0 - lse;
  out[(t + 256) * NCLASS + c] = v1 - lse;
}

// ---------------------------------------------------------------- driver
extern "C" void kernel_launch(void* const* d_in, const int* in_sizes, int n_in,
                              void* d_out, int out_size, void* d_ws, size_t ws_size,
                              hipStream_t stream) {
  (void)in_sizes; (void)n_in; (void)out_size; (void)ws_size;
  const float* h      = (const float*)d_in[0];
  const float* Ws     = (const float*)d_in[1];
  const float* bs     = (const float*)d_in[2];
  const float* gammas = (const float*)d_in[3];
  const float* betas  = (const float*)d_in[4];
  const float* fc1_w  = (const float*)d_in[5];
  const float* fc1_b  = (const float*)d_in[6];
  const float* fc2_w  = (const float*)d_in[7];
  const float* fc2_b  = (const float*)d_in[8];
  const int*   src    = (const int*)d_in[9];
  const int*   dst    = (const int*)d_in[10];
  const int*   gids   = (const int*)d_in[11];
  float* out = (float*)d_out;

  char* ws = (char*)d_ws;
  size_t off = 0;
  auto carve = [&](size_t bytes) -> void* {
    void* p = ws + off;
    off += (bytes + 255) & ~(size_t)255;
    return p;
  };
  float*          agg    = (float*)carve((size_t)ND * 4);
  float*          z      = (float*)carve((size_t)ND * 4);
  float*          h_cur  = (float*)carve((size_t)ND * 4);
  unsigned short* xb     = (unsigned short*)carve((size_t)ND * 2);
  unsigned short* Wp     = (unsigned short*)carve((size_t)5 * MATSZ * 2);
  float*          stats  = (float*)carve(2 * D * 4);
  float*          pool   = (float*)carve((size_t)NGRAPH * D * 4);
  float*          counts = (float*)carve(NGRAPH * 4);
  unsigned short* hgb    = (unsigned short*)carve((size_t)NGRAPH * D * 2);
  float*          zfc    = (float*)carve((size_t)NGRAPH * D * 4);
  float*          hgelu  = (float*)carve((size_t)NGRAPH * D * 4);
  float*          logits = (float*)carve((size_t)NGRAPH * NCLASS * 4);

  // Pack all bf16 weight fragments once per call.
  pack_weights_kernel<<<(5 * MATSZ + 255) / 256, 256, 0, stream>>>(Ws, fc1_w, Wp);

  const float* hin = h;
  for (int l = 0; l < 4; ++l) {
    fill_kernel<<<(ND + 255) / 256, 256, 0, stream>>>(agg, ND);
    fill_kernel<<<2, 256, 0, stream>>>(stats, 2 * D);
    scatter_kernel<<<(size_t)NEDGES * 64 / 256, 256, 0, stream>>>(hin, src, dst, agg);
    addcvt_kernel<<<ND / 256, 256, 0, stream>>>(hin, agg, xb, ND);
    wmma_gemm_kernel<<<(NNODES + 63) / 64, 128, 0, stream>>>(
        xb, Wp + (size_t)l * MATSZ, bs + l * D, z, NNODES);
    colstats_kernel<<<(NNODES + 255) / 256, 256, 0, stream>>>(z, stats, NNODES);
    bnrelu_kernel<<<ND / 256, 256, 0, stream>>>(
        z, stats, gammas + l * D, betas + l * D, h_cur, ND);
    hin = h_cur;
  }

  // Average pooling per graph.
  fill_kernel<<<(NGRAPH * D + 255) / 256, 256, 0, stream>>>(pool, NGRAPH * D);
  fill_kernel<<<2, 256, 0, stream>>>(counts, NGRAPH);
  count_kernel<<<(NNODES + 255) / 256, 256, 0, stream>>>(gids, counts, NNODES);
  poolsum_kernel<<<ND / 256, 256, 0, stream>>>(h_cur, gids, pool, ND);
  poolfin_kernel<<<NGRAPH * D / 256, 256, 0, stream>>>(pool, counts, hgb);

  // fc1 (WMMA) + ELU, fc2, log_softmax(dim=0).
  wmma_gemm_kernel<<<NGRAPH / 64, 128, 0, stream>>>(hgb, Wp + (size_t)4 * MATSZ,
                                                    fc1_b, zfc, NGRAPH);
  elu_kernel<<<NGRAPH * D / 256, 256, 0, stream>>>(zfc, hgelu, NGRAPH * D);
  fc2_kernel<<<(NGRAPH * NCLASS + 255) / 256, 256, 0, stream>>>(hgelu, fc2_w,
                                                                fc2_b, logits);
  logsoftmax_kernel<<<NCLASS, 256, 0, stream>>>(logits, out);
}
